// SpatialGatherModule_61340722922142
// MI455X (gfx1250) — matrix operand order repaired and checked by reading, not compiled, not run on hardware
//
#include <hip/hip_runtime.h>
#include <hip/hip_bf16.h>
#include <math.h>

// Problem constants (from reference)
#define BB 8
#define CC 512
#define KK 19
#define SS 16384          // H*W
#define KP 32             // K rows padded to 2 WMMA m-tiles (rows 19..31 zero)

// Tiling
#define SC    4096        // s elements per chunk (per block)
#define NSCH  (SS / SC)   // 4
#define CTILE 32          // output columns per block (2 WMMA n-tiles)
#define NCT   (CC / CTILE)// 16
#define SBLK  256         // s elements staged per round
#define NROUND (SC / SBLK)// 16
#define AROWP 264         // bf16 elems/row; 528B stride (bank-spread)
#define BROWP 260         // f32  elems/row; 1040B stride (bank-spread)

#define P_OFF_BYTES 4096  // P matrix offset inside d_ws (stats live below)

typedef __attribute__((ext_vector_type(16))) __bf16 v16bf;
typedef __attribute__((ext_vector_type(8)))  __bf16 v8bf;
typedef __attribute__((ext_vector_type(4)))  __bf16 v4bf;
typedef __attribute__((ext_vector_type(8)))  float  v8f;
typedef __attribute__((ext_vector_type(4)))  float  v4f;
typedef __attribute__((ext_vector_type(4)))  int    v4i;

// ---------------- CDNA5 async global->LDS probe ----------------
#if defined(__has_builtin)
#if __has_builtin(__builtin_amdgcn_global_load_async_to_lds_b128) && \
    __has_builtin(__builtin_amdgcn_s_wait_asynccnt)
#define HAS_ASYNC 1
#endif
#endif
#ifndef HAS_ASYNC
#define HAS_ASYNC 0
#endif

#if HAS_ASYNC
#define GAS __attribute__((address_space(1)))
#define LAS __attribute__((address_space(3)))
// Param 1 is a non-const global v4i*; param 2 an LDS v4i* (per round-2 diagnostics).
#define ASYNC_COPY16(g, l) \
    __builtin_amdgcn_global_load_async_to_lds_b128((GAS v4i*)(g), (LAS v4i*)(l), 0, 0)
#define WAIT_ASYNC(n) __builtin_amdgcn_s_wait_asynccnt(n)
#else
#define ASYNC_COPY16(g, l) (*(v4i*)(l) = *(const v4i*)(g))
#define WAIT_ASYNC(n) ((void)0)
#endif

__device__ __forceinline__ v16bf combine16(v8bf lo, v8bf hi) {
    v16bf r;
#pragma unroll
    for (int e = 0; e < 8; ++e) { r[e] = lo[e]; r[8 + e] = hi[e]; }
    return r;
}

// 16 consecutive f32 from LDS -> bf16 fragment (v_cvt_pk_bf16_f32 in regs)
__device__ __forceinline__ v16bf cvt16(const float* p) {
    v16bf r;
#pragma unroll
    for (int e = 0; e < 16; ++e) r[e] = (__bf16)p[e];
    return r;
}

// ---------------- Pass 1: per-row softmax stats (max, 1/sum) ----------------
__global__ void __launch_bounds__(256)
sg_rowstats(const float* __restrict__ probs, float* __restrict__ ws) {
    __shared__ float red[256];
    const int r = blockIdx.x;                 // r = b*KK + k
    const float* row = probs + (size_t)r * SS;
    const int tid = threadIdx.x;

    float m = -INFINITY;
    for (int i = tid; i < SS; i += 256) m = fmaxf(m, row[i]);
    red[tid] = m; __syncthreads();
    for (int off = 128; off > 0; off >>= 1) {
        if (tid < off) red[tid] = fmaxf(red[tid], red[tid + off]);
        __syncthreads();
    }
    m = red[0]; __syncthreads();

    float s = 0.f;
    for (int i = tid; i < SS; i += 256) s += __expf(row[i] - m);
    red[tid] = s; __syncthreads();
    for (int off = 128; off > 0; off >>= 1) {
        if (tid < off) red[tid] += red[tid + off];
        __syncthreads();
    }
    if (tid == 0) { ws[r] = m; ws[BB * KK + r] = 1.0f / red[0]; }
}

// ------- Pass 1.5: P[b, 32, S] = bf16(softmax), rows 19..31 zeroed ---------
__global__ void __launch_bounds__(256)
sg_softmax(const float* __restrict__ probs, const float* __restrict__ ws,
           unsigned short* __restrict__ Praw) {
    __bf16* P = (__bf16*)Praw;
    const int row = blockIdx.x;               // b*KP + r
    const int b = row / KP, r = row % KP;
    __bf16* dst = P + (size_t)row * SS;
    const int tid = threadIdx.x;

    if (r >= KK) {                            // zero pad rows
        for (int i = tid * 8; i < SS; i += 256 * 8)
            *(v4i*)(dst + i) = (v4i){0, 0, 0, 0};
        return;
    }
    const float m = ws[b * KK + r];
    const float z = ws[BB * KK + b * KK + r];
    const float* src = probs + ((size_t)b * KK + r) * SS;
    for (int i = tid * 4; i < SS; i += 256 * 4) {
        const v4f x = *(const v4f*)(src + i);
        v4bf h;
        h.x = (__bf16)(__expf(x.x - m) * z);
        h.y = (__bf16)(__expf(x.y - m) * z);
        h.z = (__bf16)(__expf(x.z - m) * z);
        h.w = (__bf16)(__expf(x.w - m) * z);
        *(v4bf*)(dst + i) = h;
    }
}

// ---------------- Pass 2: double-buffered async WMMA GEMM ------------------
template <bool USE_P>
__global__ void __launch_bounds__(256)
sg_gemm(const float* __restrict__ probs, const float* __restrict__ features,
        const float* __restrict__ ws, const unsigned short* __restrict__ Praw,
        float* __restrict__ out) {
    __shared__ __bf16 Abuf[2][KP * AROWP];    // softmax weights (bf16)
    __shared__ float  Bbuf[2][CTILE * BROWP]; // features (raw f32, cvt at frag load)
    __shared__ float  Acc[KP * CTILE];
    __shared__ float  sM[KK], sI[KK];

    const int tid = threadIdx.x;
    const int bid = blockIdx.x;
    const int sch = bid & (NSCH - 1);
    const int ct  = (bid / NSCH) & (NCT - 1);
    const int b   = bid / (NSCH * NCT);
    const int c0  = ct * CTILE;
    const int s0  = sch * SC;

    if (tid < KK) { sM[tid] = ws[b * KK + tid]; sI[tid] = ws[BB * KK + b * KK + tid]; }
    for (int i = tid; i < KP * CTILE; i += 256) Acc[i] = 0.f;
    if (!USE_P) {   // inline-exp path writes only rows 0..18: zero pads once
        for (int i = tid; i < (KP - KK) * AROWP; i += 256) {
            Abuf[0][KK * AROWP + i] = (__bf16)0.f;
            Abuf[1][KK * AROWP + i] = (__bf16)0.f;
        }
    }
    __syncthreads();   // sM/sI visible before staging

    const float*  fbase = features + ((size_t)b * CC + c0) * SS + s0;
    const float*  pbase = probs + ((size_t)b * KK) * SS + s0;
    const __bf16* pb    = (const __bf16*)Praw + (size_t)b * KP * SS + s0;

    // Stage one 256-wide s round into LDS buffer (rd & 1).
    // Async-issue count is uniform per wave: 8 (B) + 4 (A, USE_P only).
    auto stage = [&](int rd) {
        const int buf = rd & 1;
        const int sb  = rd * SBLK;
#pragma unroll
        for (int it = 0; it < 8; ++it) {       // B: 32 rows x 256 f32, 16B chunks
            const int idx = tid + it * 256;
            const int row = idx >> 6;
            const int ch  = idx & 63;
            ASYNC_COPY16(fbase + (size_t)row * SS + sb + ch * 4,
                         &Bbuf[buf][row * BROWP + ch * 4]);
        }
        if (USE_P) {
#pragma unroll
            for (int it = 0; it < 4; ++it) {   // A: 32 rows x 256 bf16, 16B chunks
                const int idx = tid + it * 256;
                const int row = idx >> 5;
                const int ch  = idx & 31;
                ASYNC_COPY16(pb + (size_t)row * SS + sb + ch * 8,
                             &Abuf[buf][row * AROWP + ch * 8]);
            }
        } else {                               // inline softmax -> bf16
            for (int idx = tid; idx < KK * SBLK; idx += 256) {
                const int k = idx >> 8;
                const int i = idx & (SBLK - 1);
                const float x = pbase[(size_t)k * SS + sb + i];
                Abuf[buf][k * AROWP + i] = (__bf16)(__expf(x - sM[k]) * sI[k]);
            }
        }
    };

    const int wave  = tid >> 5;
    const int lane  = tid & 31;
    const int lhalf = (lane < 16) ? 0 : 1;
    const int l16   = lane & 15;

    v8f acc00 = {}, acc01 = {}, acc10 = {}, acc11 = {};

    stage(0);
    for (int rd = 0; rd < NROUND; ++rd) {
        if (rd + 1 < NROUND) {
            stage(rd + 1);                     // prefetch next round into other buffer
            if (USE_P) WAIT_ASYNC(12); else WAIT_ASYNC(8);  // drain round rd only
        } else {
            WAIT_ASYNC(0);
        }
        __syncthreads();                       // all waves' round-rd data in LDS

        const __bf16* Ab = Abuf[rd & 1];
        const float*  Bb = Bbuf[rd & 1];
        const int sB = wave * 32;              // this wave's 32-wide K slice

        // A frags: lanes<16 hold K0..15; lanes>=16 K16..31
        const int a0r = l16, a1r = 16 + l16;
        const v16bf a0 = combine16(*(const v8bf*)(Ab + a0r * AROWP + sB + lhalf * 16),
                                   *(const v8bf*)(Ab + a0r * AROWP + sB + 8 + lhalf * 16));
        const v16bf a1 = combine16(*(const v8bf*)(Ab + a1r * AROWP + sB + lhalf * 16),
                                   *(const v8bf*)(Ab + a1r * AROWP + sB + 8 + lhalf * 16));
        // B frags: lane = column N, 16 consecutive K (f32 -> bf16 in regs)
        const int kb = lhalf * 16;
        const v16bf b0 = cvt16(Bb + l16 * BROWP + sB + kb);
        const v16bf b1 = cvt16(Bb + (16 + l16) * BROWP + sB + kb);

        acc00 = __builtin_amdgcn_wmma_f32_16x16x32_bf16(false, a0, false, b0, (short)0, acc00, false, false);
        acc01 = __builtin_amdgcn_wmma_f32_16x16x32_bf16(false, a0, false, b1, (short)0, acc01, false, false);
        acc10 = __builtin_amdgcn_wmma_f32_16x16x32_bf16(false, a1, false, b0, (short)0, acc10, false, false);
        acc11 = __builtin_amdgcn_wmma_f32_16x16x32_bf16(false, a1, false, b1, (short)0, acc11, false, false);

        __syncthreads();                       // safe to overwrite buf (rd-1)
    }

    // cross-wave reduction in LDS (ds_add_f32)
#pragma unroll
    for (int r = 0; r < 8; ++r) {
        const int mrow = r + lhalf * 8;
        atomicAdd(&Acc[mrow * CTILE + l16],             acc00[r]);
        atomicAdd(&Acc[mrow * CTILE + 16 + l16],        acc01[r]);
        atomicAdd(&Acc[(16 + mrow) * CTILE + l16],      acc10[r]);
        atomicAdd(&Acc[(16 + mrow) * CTILE + 16 + l16], acc11[r]);
    }
    __syncthreads();

    // accumulate this block's partial into global output (global_atomic_add_f32)
    for (int idx = tid; idx < KK * CTILE; idx += 256) {
        const int k = idx / CTILE;
        const int c = idx % CTILE;
        unsafeAtomicAdd(&out[((size_t)b * KK + k) * CC + c0 + c], Acc[k * CTILE + c]);
    }
}

// ---------------- host-side launcher ----------------
extern "C" void kernel_launch(void* const* d_in, const int* in_sizes, int n_in,
                              void* d_out, int out_size, void* d_ws, size_t ws_size,
                              hipStream_t stream) {
    const float* features = (const float*)d_in[0]; // [B, C, H, W] f32
    const float* probs    = (const float*)d_in[1]; // [B, K, H, W] f32
    float* out = (float*)d_out;                    // [B, K, C] f32
    float* ws  = (float*)d_ws;
    unsigned short* P = (unsigned short*)((char*)d_ws + P_OFF_BYTES);

    (void)in_sizes; (void)n_in;

    const size_t needWs = (size_t)P_OFF_BYTES + (size_t)BB * KP * SS * 2u;
    const bool useP = (ws_size >= needWs);     // fixed per run -> deterministic

    (void)hipMemsetAsync(d_out, 0, (size_t)out_size * sizeof(float), stream);
    sg_rowstats<<<BB * KK, 256, 0, stream>>>(probs, ws);
    if (useP) {
        sg_softmax<<<BB * KP, 256, 0, stream>>>(probs, ws, P);
        sg_gemm<true><<<BB * NCT * NSCH, 256, 0, stream>>>(probs, features, ws, P, out);
    } else {
        sg_gemm<false><<<BB * NCT * NSCH, 256, 0, stream>>>(probs, features, ws, nullptr, out);
    }
}
